// PluggableMHA_8770323218614
// MI455X (gfx1250) — matrix operand order repaired and checked by reading, not compiled, not run on hardware
//
#include <hip/hip_runtime.h>

// ---------------------------------------------------------------------------
// PluggableMHA (dense-synthesizer + vanilla attention) for MI455X / gfx1250.
// bf16 WMMA (v_wmma_f32_16x16x32_bf16) everywhere; wave32; 8-wave blocks.
// Double-buffered GEMM staging via GLOBAL_LOAD_ASYNC_TO_LDS (ASYNCcnt).
// ---------------------------------------------------------------------------

typedef __bf16 bf16;
typedef __attribute__((ext_vector_type(16))) __bf16 v16bf;
typedef __attribute__((ext_vector_type(8)))  float  v8f;
typedef __attribute__((ext_vector_type(4)))  int    v4i;

#define D_MODEL 1024
#define N_HEADS 16
#define D_HEAD  64
#define TQ_     1024
#define TK_     1024
#define B_      2
#define NTOT    16384   // N_HEADS * MAX_K_LEN

#if __has_builtin(__builtin_amdgcn_global_load_async_to_lds_b128) && \
    __has_builtin(__builtin_amdgcn_s_wait_asynccnt)
#define ASYNC_LDS 1
#else
#define ASYNC_LDS 0
#endif

// Builtin expects v4i pointers: AS(1) for global source, AS(3) for LDS dest.
typedef __attribute__((address_space(1))) v4i* gv4i_p;
typedef __attribute__((address_space(3))) v4i* lv4i_p;

// Generic->AS casts via integers (LDS generic addr low 32 bits == LDS offset).
__device__ __forceinline__ gv4i_p as_gbl(const void* p) {
    return (gv4i_p)(unsigned long long)(uintptr_t)p;
}
__device__ __forceinline__ lv4i_p as_lds(void* p) {
    return (lv4i_p)(unsigned int)(uintptr_t)p;
}

union FragAB {
    v16bf v;
    uint4 q[2];
    bf16  h[16];
};

__device__ __forceinline__ v8f wmma_bf16(const FragAB& a, const FragAB& b, v8f c) {
    return __builtin_amdgcn_wmma_f32_16x16x32_bf16(false, a.v, false, b.v,
                                                   (short)0, c, false, false);
}

__device__ __forceinline__ v8f v8f_zero() {
    v8f z = {0.f, 0.f, 0.f, 0.f, 0.f, 0.f, 0.f, 0.f};
    return z;
}

// copy 16 bf16 (32B) global -> LDS, async if available
__device__ __forceinline__ void stage16(const bf16* g, bf16* l) {
#if ASYNC_LDS
    __builtin_amdgcn_global_load_async_to_lds_b128(as_gbl(g),     as_lds(l),     0, 0);
    __builtin_amdgcn_global_load_async_to_lds_b128(as_gbl(g + 8), as_lds(l + 8), 0, 0);
#else
    ((uint4*)l)[0] = ((const uint4*)g)[0];
    ((uint4*)l)[1] = ((const uint4*)g)[1];
#endif
}

__device__ __forceinline__ void wait_async() {
#if ASYNC_LDS
    __builtin_amdgcn_s_wait_asynccnt(0);
#endif
}

// -------------------------- f32 -> bf16 convert ----------------------------
__global__ __launch_bounds__(256) void cvt_f32_bf16(const float* __restrict__ src,
                                                    bf16* __restrict__ dst, int n) {
    int i = blockIdx.x * 256 + threadIdx.x;
    if (i < n) dst[i] = (bf16)src[i];
}

// ------------------ transpose + convert: dst[C][R] = src[R][C] -------------
__global__ __launch_bounds__(256) void transpose_f32_bf16(const float* __restrict__ src,
                                                          bf16* __restrict__ dst,
                                                          int R, int C) {
    __shared__ float t[32][33];
    const int c0 = blockIdx.x * 32, r0 = blockIdx.y * 32;
    const int x = threadIdx.x & 31, y = threadIdx.x >> 5;   // 32 x 8
    #pragma unroll
    for (int i = 0; i < 32; i += 8)
        t[y + i][x] = src[(size_t)(r0 + y + i) * C + c0 + x];
    __syncthreads();
    #pragma unroll
    for (int i = 0; i < 32; i += 8)
        dst[(size_t)(c0 + y + i) * R + r0 + x] = (bf16)t[x][y + i];
}

// ------------------------- WMMA GEMM: C = A @ Bt^T + bias ------------------
// A  : [M][K]  bf16 row-major
// Bt : [N][K]  bf16 (transposed weight -> B-operand rows stage directly)
// ACT: 0 = none, 1 = exact GELU.  OUT_BF16: 1 = bf16 out, 0 = f32 out.
// Double-buffered LDS; next tile fetched asynchronously while computing.
template <int ACT, int OUT_BF16>
__global__ __launch_bounds__(256) void gemm_bf16_wmma(
        const bf16* __restrict__ A, const bf16* __restrict__ Bt,
        const float* __restrict__ bias, void* __restrict__ Cout,
        int M, int N, int K) {
    __shared__ bf16 Alds[2][128 * 40];   // 32 K halves + 8 pad per row
    __shared__ bf16 Blds[2][128 * 40];

    const int tid  = threadIdx.x;
    const int lane = tid & 31, wid = tid >> 5;
    const int lh = lane & 15, sel = lane >> 4;
    const int mBase = blockIdx.y * 128, nBase = blockIdx.x * 128;

    // staging: thread -> (row 0..127, 16-half chunk 0/16)
    const int srow = tid >> 1, schunk = (tid & 1) * 16;
    const bf16* gA = A  + (size_t)(mBase + srow) * K + schunk;
    const bf16* gB = Bt + (size_t)(nBase + srow) * K + schunk;
    const int soff = srow * 40 + schunk;

    v8f acc[8];
    #pragma unroll
    for (int j = 0; j < 8; ++j) acc[j] = v8f_zero();

    const int aRow = wid * 16 + lh;
    const int c0 = sel ? 8 : 0;       // A-frag K sub-chunk (ISA 16-bit A layout)
    const int kk = sel ? 8 : 0;       // B-frag K sub-chunk (same order as A)

    // prologue: stage K-step 0 into buffer 0
    stage16(gA, &Alds[0][soff]);
    stage16(gB, &Blds[0][soff]);

    int cur = 0;
    for (int k0 = 0; k0 < K; k0 += 32, cur ^= 1) {
        wait_async();          // this wave's loads of buffer `cur` complete
        __syncthreads();       // whole WG: buffer `cur` ready, buffer `cur^1` free
        if (k0 + 32 < K) {     // overlap next tile fetch with compute
            stage16(gA + k0 + 32, &Alds[cur ^ 1][soff]);
            stage16(gB + k0 + 32, &Blds[cur ^ 1][soff]);
        }

        FragAB af;
        af.q[0] = *(const uint4*)&Alds[cur][aRow * 40 + c0];
        af.q[1] = *(const uint4*)&Alds[cur][aRow * 40 + c0 + 16];
        #pragma unroll
        for (int j = 0; j < 8; ++j) {
            FragAB bfr;
            bfr.q[0] = *(const uint4*)&Blds[cur][(j * 16 + lh) * 40 + kk];
            bfr.q[1] = *(const uint4*)&Blds[cur][(j * 16 + lh) * 40 + kk + 16];
            acc[j] = wmma_bf16(af, bfr, acc[j]);
        }
    }

    // epilogue: C layout m = wid*16 + 8*sel + r, n = nBase + j*16 + lh
    const int mRow0 = mBase + wid * 16 + sel * 8;
    #pragma unroll
    for (int j = 0; j < 8; ++j) {
        const int n = nBase + j * 16 + lh;
        const float bv = bias[n];
        #pragma unroll
        for (int r = 0; r < 8; ++r) {
            float v = acc[j][r] + bv;
            if (ACT == 1) v = 0.5f * v * (1.0f + erff(v * 0.70710678118f));
            const size_t oi = (size_t)(mRow0 + r) * N + n;
            if (OUT_BF16) ((bf16*)Cout)[oi] = (bf16)v;
            else          ((float*)Cout)[oi] = v;
        }
    }
}

// --------------------- flash attention + synthesizer gate ------------------
// grid: (TQ/128, H, B). block 256 = 8 waves, each wave owns 16 q rows.
__global__ __launch_bounds__(256) void attn_kernel(
        const bf16* __restrict__ Qb, const bf16* __restrict__ Kb,
        const bf16* __restrict__ Vb, const bf16* __restrict__ Sb,
        const int* __restrict__ kpm, const float* __restrict__ gate,
        bf16* __restrict__ ctxb) {
    __shared__ bf16 Klds[64 * 72];        // K tile, row-major [k][d]
    __shared__ bf16 Vtlds[64 * 72];       // V tile transposed [d][k]
    __shared__ bf16 Plds[8][16 * 72];     // per-wave P tile round-trip

    const int tid  = threadIdx.x;
    const int lane = tid & 31, wid = tid >> 5;
    const int lh = lane & 15, sel = lane >> 4;
    const int qBase = blockIdx.x * 128;
    const int h = blockIdx.y;
    const int b = blockIdx.z;

    const float g  = 1.0f / (1.0f + expf(-gate[h]));
    const float og = (1.0f - g) * 0.125f;          // (1-g) * dh^-0.5

    const int c0 = sel ? 8 : 0, kksel = sel ? 8 : 0;

    // Q fragments live in registers across the whole k loop
    FragAB qa[2];
    {
        const size_t rowQ = (size_t)b * TQ_ + qBase + wid * 16 + lh;
        #pragma unroll
        for (int c = 0; c < 2; ++c) {
            const bf16* p = Qb + rowQ * D_MODEL + h * D_HEAD + c * 32 + c0;
            qa[c].q[0] = *(const uint4*)p;
            qa[c].q[1] = *(const uint4*)(p + 16);
        }
    }

    v8f acc[4];
    #pragma unroll
    for (int j = 0; j < 4; ++j) acc[j] = v8f_zero();
    float mrow[8], lsum[8];
    #pragma unroll
    for (int r = 0; r < 8; ++r) { mrow[r] = -3.0e38f; lsum[r] = 0.0f; }

    const int srow = tid >> 2, schunk = (tid & 3) * 16;   // staging map
    const int kEnd = qBase + 128;                          // causal bound
    // synth logits row base for this lane's 8 C-layout rows
    const size_t sRow0 = ((size_t)b * TQ_ + qBase + wid * 16 + sel * 8) * NTOT
                       + (size_t)h * 1024;

    for (int kt = 0; kt < kEnd; kt += 64) {
        __syncthreads();
        // stage K tile (row-major) -- async copy into LDS
        stage16(Kb + ((size_t)b * TK_ + kt + srow) * D_MODEL + h * D_HEAD + schunk,
                &Klds[srow * 72 + schunk]);
        {   // stage V tile transposed -> Vt[d][k] (through registers)
            FragAB u;
            const bf16* p = Vb + ((size_t)b * TK_ + kt + srow) * D_MODEL
                          + h * D_HEAD + schunk;
            u.q[0] = ((const uint4*)p)[0];
            u.q[1] = ((const uint4*)p)[1];
            #pragma unroll
            for (int i = 0; i < 16; ++i)
                Vtlds[(schunk + i) * 72 + srow] = u.h[i];
        }
        wait_async();
        __syncthreads();

        // vanilla logits: QK^T (dh=64 -> two K=32 WMMA steps)
        v8f lg[4];
        #pragma unroll
        for (int j = 0; j < 4; ++j) {
            lg[j] = v8f_zero();
            #pragma unroll
            for (int c = 0; c < 2; ++c) {
                FragAB bfr;
                bfr.q[0] = *(const uint4*)&Klds[(j * 16 + lh) * 72 + c * 32 + kksel];
                bfr.q[1] = *(const uint4*)&Klds[(j * 16 + lh) * 72 + c * 32 + kksel + 16];
                lg[j] = wmma_bf16(qa[c], bfr, lg[j]);
            }
        }

        // gate with synthesizer logits; branchless causal + padding masks
        #pragma unroll
        for (int j = 0; j < 4; ++j) {
            const int k_idx = kt + j * 16 + lh;
            const int pad   = kpm[b * TK_ + k_idx];
            #pragma unroll
            for (int r = 0; r < 8; ++r) {
                const int q_idx = qBase + wid * 16 + sel * 8 + r;
                const float sv  = (float)Sb[sRow0 + (size_t)r * NTOT + k_idx];
                const bool keep = (pad > 0) && (k_idx <= q_idx);
                lg[j][r] = keep ? (og * lg[j][r] + g * sv) : -3.0e38f;
            }
        }

        // online softmax: rows live across the 16-lane half (C layout)
        float rmax[8];
        #pragma unroll
        for (int r = 0; r < 8; ++r) {
            float m = lg[0][r];
            m = fmaxf(m, lg[1][r]); m = fmaxf(m, lg[2][r]); m = fmaxf(m, lg[3][r]);
            rmax[r] = m;
        }
        #pragma unroll
        for (int off = 1; off < 16; off <<= 1)
            #pragma unroll
            for (int r = 0; r < 8; ++r)
                rmax[r] = fmaxf(rmax[r], __shfl_xor(rmax[r], off, 16));

        float alpha[8];
        #pragma unroll
        for (int r = 0; r < 8; ++r) {
            const float mn = fmaxf(mrow[r], rmax[r]);
            alpha[r] = expf(mrow[r] - mn);
            mrow[r] = mn;
        }
        #pragma unroll
        for (int j = 0; j < 4; ++j)
            #pragma unroll
            for (int r = 0; r < 8; ++r) acc[j][r] *= alpha[r];

        float rsum[8];
        #pragma unroll
        for (int r = 0; r < 8; ++r) rsum[r] = 0.0f;
        #pragma unroll
        for (int j = 0; j < 4; ++j)
            #pragma unroll
            for (int r = 0; r < 8; ++r) {
                const float p = expf(lg[j][r] - mrow[r]);
                rsum[r] += p;
                Plds[wid][(sel * 8 + r) * 72 + j * 16 + lh] = (bf16)p;
            }
        #pragma unroll
        for (int off = 1; off < 16; off <<= 1)
            #pragma unroll
            for (int r = 0; r < 8; ++r)
                rsum[r] += __shfl_xor(rsum[r], off, 16);
        #pragma unroll
        for (int r = 0; r < 8; ++r) lsum[r] = lsum[r] * alpha[r] + rsum[r];

        // ctx += P @ V  (k=64 -> two K=32 WMMA steps; wave-local LDS round-trip)
        #pragma unroll
        for (int c = 0; c < 2; ++c) {
            FragAB pa;
            pa.q[0] = *(const uint4*)&Plds[wid][lh * 72 + c * 32 + c0];
            pa.q[1] = *(const uint4*)&Plds[wid][lh * 72 + c * 32 + c0 + 16];
            #pragma unroll
            for (int j = 0; j < 4; ++j) {
                FragAB vb;
                vb.q[0] = *(const uint4*)&Vtlds[(j * 16 + lh) * 72 + c * 32 + kksel];
                vb.q[1] = *(const uint4*)&Vtlds[(j * 16 + lh) * 72 + c * 32 + kksel + 16];
                acc[j] = wmma_bf16(pa, vb, acc[j]);
            }
        }
    }

    // normalize and emit bf16 ctx [b][q][h*64+d]
    #pragma unroll
    for (int r = 0; r < 8; ++r) {
        const float inv = 1.0f / lsum[r];
        const int q_idx = qBase + wid * 16 + sel * 8 + r;
        #pragma unroll
        for (int j = 0; j < 4; ++j) {
            const float o = acc[j][r] * inv;
            ctxb[((size_t)b * TQ_ + q_idx) * D_MODEL + h * D_HEAD + j * 16 + lh] = (bf16)o;
        }
    }
}

// ---------------------------------------------------------------------------
extern "C" void kernel_launch(void* const* d_in, const int* in_sizes, int n_in,
                              void* d_out, int out_size, void* d_ws, size_t ws_size,
                              hipStream_t stream) {
    (void)in_sizes; (void)n_in; (void)out_size; (void)ws_size;

    const float* x_q  = (const float*)d_in[0];
    const float* x_kv = (const float*)d_in[1];
    const int*   kpm  = (const int*)  d_in[2];
    const float* Wq   = (const float*)d_in[3];
    const float* bq   = (const float*)d_in[4];
    const float* Wk   = (const float*)d_in[5];
    const float* bk   = (const float*)d_in[6];
    const float* Wv   = (const float*)d_in[7];
    const float* bv   = (const float*)d_in[8];
    const float* Wo   = (const float*)d_in[9];
    const float* bo   = (const float*)d_in[10];
    const float* Ws1  = (const float*)d_in[11];
    const float* bs1  = (const float*)d_in[12];
    const float* Ws2  = (const float*)d_in[13];
    const float* bs2  = (const float*)d_in[14];
    const float* gate = (const float*)d_in[15];

    const size_t R = (size_t)B_ * TQ_;         // 2048 token rows
    char* w = (char*)d_ws;
    bf16* xq_b  = (bf16*)w;  w += R * D_MODEL * 2;
    bf16* xkv_b = (bf16*)w;  w += R * D_MODEL * 2;
    bf16* Wqt   = (bf16*)w;  w += (size_t)D_MODEL * D_MODEL * 2;
    bf16* Wkt   = (bf16*)w;  w += (size_t)D_MODEL * D_MODEL * 2;
    bf16* Wvt   = (bf16*)w;  w += (size_t)D_MODEL * D_MODEL * 2;
    bf16* Wot   = (bf16*)w;  w += (size_t)D_MODEL * D_MODEL * 2;
    bf16* Ws1t  = (bf16*)w;  w += (size_t)D_MODEL * D_MODEL * 2;
    bf16* Ws2t  = (bf16*)w;  w += (size_t)NTOT * D_MODEL * 2;   // [16384][1024]
    bf16* Qb    = (bf16*)w;  w += R * D_MODEL * 2;
    bf16* Kb    = (bf16*)w;  w += R * D_MODEL * 2;
    bf16* Vb    = (bf16*)w;  w += R * D_MODEL * 2;
    bf16* hb    = (bf16*)w;  w += R * D_MODEL * 2;
    bf16* Sb    = (bf16*)w;  w += R * (size_t)NTOT * 2;         // 64 MB
    bf16* ctxb  = (bf16*)w;  w += R * D_MODEL * 2;

    const int nAct = (int)(R * D_MODEL);
    cvt_f32_bf16<<<(nAct + 255) / 256, 256, 0, stream>>>(x_q,  xq_b,  nAct);
    cvt_f32_bf16<<<(nAct + 255) / 256, 256, 0, stream>>>(x_kv, xkv_b, nAct);

    transpose_f32_bf16<<<dim3(32, 32),  256, 0, stream>>>(Wq,  Wqt,  D_MODEL, D_MODEL);
    transpose_f32_bf16<<<dim3(32, 32),  256, 0, stream>>>(Wk,  Wkt,  D_MODEL, D_MODEL);
    transpose_f32_bf16<<<dim3(32, 32),  256, 0, stream>>>(Wv,  Wvt,  D_MODEL, D_MODEL);
    transpose_f32_bf16<<<dim3(32, 32),  256, 0, stream>>>(Wo,  Wot,  D_MODEL, D_MODEL);
    transpose_f32_bf16<<<dim3(32, 32),  256, 0, stream>>>(Ws1, Ws1t, D_MODEL, D_MODEL);
    transpose_f32_bf16<<<dim3(512, 32), 256, 0, stream>>>(Ws2, Ws2t, D_MODEL, NTOT);

    // Q/K/V projections (bf16 out)
    gemm_bf16_wmma<0, 1><<<dim3(8, 16), 256, 0, stream>>>(xq_b,  Wqt,  bq,  Qb,
                                                          (int)R, D_MODEL, D_MODEL);
    gemm_bf16_wmma<0, 1><<<dim3(8, 16), 256, 0, stream>>>(xkv_b, Wkt,  bk,  Kb,
                                                          (int)R, D_MODEL, D_MODEL);
    gemm_bf16_wmma<0, 1><<<dim3(8, 16), 256, 0, stream>>>(xkv_b, Wvt,  bv,  Vb,
                                                          (int)R, D_MODEL, D_MODEL);
    // synthesizer MLP: h = gelu(x_q @ Ws1 + bs1); S = h @ Ws2 + bs2
    gemm_bf16_wmma<1, 1><<<dim3(8, 16), 256, 0, stream>>>(xq_b, Ws1t, bs1, hb,
                                                          (int)R, D_MODEL, D_MODEL);
    gemm_bf16_wmma<0, 1><<<dim3(128, 16), 256, 0, stream>>>(hb, Ws2t, bs2, Sb,
                                                            (int)R, NTOT, D_MODEL);
    // gated flash attention
    attn_kernel<<<dim3(TQ_ / 128, N_HEADS, B_), 256, 0, stream>>>(
        Qb, Kb, Vb, Sb, kpm, gate, ctxb);
    // output projection (f32 out to d_out)
    gemm_bf16_wmma<0, 0><<<dim3(8, 16), 256, 0, stream>>>(ctxb, Wot, bo, d_out,
                                                          (int)R, D_MODEL, D_MODEL);
}
